// GNNEncoder_45973329937095
// MI455X (gfx1250) — compile-verified, matrix operand order checked by reading, not run on hardware
//
#include <hip/hip_runtime.h>

typedef __attribute__((ext_vector_type(16))) _Float16 v16h;
typedef __attribute__((ext_vector_type(8)))  float    v8f;

// ---------------------------------------------------------------- utilities
__global__ void fill_kernel(float* __restrict__ p, float v, int n) {
  int i = blockIdx.x * blockDim.x + threadIdx.x;
  if (i < n) p[i] = v;
}

__global__ void degree_kernel(const int* __restrict__ ei, float* __restrict__ deg, int E) {
  int e = blockIdx.x * blockDim.x + threadIdx.x;
  if (e < E) atomicAdd(&deg[ei[E + e]], 1.0f);
}

__global__ void rsqrt_kernel(float* __restrict__ deg, int n) {
  int i = blockIdx.x * blockDim.x + threadIdx.x;
  if (i < n) deg[i] = rsqrtf(deg[i]);
}

// ------------------------------------------------- WMMA GEMM: C = A @ W
// A: M x K (f32, row-major), W: K x Nc (f32, row-major).
// Nc = NT*16, K multiple of 32. Each wave computes a 32-row strip (2 M-tiles).
//
// W is staged to LDS *pre-swizzled into B-fragment order* so the hot loop
// fetches each 16-half fragment as 32 contiguous bytes (2x ds_load_b128):
//   frag(kc, t) lane l holds column t*16+(l&15); halves j=0..7 -> K=k0..k0+7,
//   j=8..15 -> K=k0+16..k0+23, where k0 = kc*32 + (l>=16 ? 8 : 0).
// Layout index: ((kc*NT + t)*32 + lane)*16 + j.
//
// Fragment layouts per CDNA5 ISA 7.12.2 (wave32):
//   A (16x32 f16): lanes 0-15 -> row M=lane, K in {0..7,16..23};
//                  lanes 16-31 -> row M=lane-16, K in {8..15,24..31}
//   C/D (16x16 f32, 8 VGPRs): vgpr i, lanes 0-15 -> M=i; lanes 16-31 -> M=8+i
template <int NT>
__global__ void wmma_gemm_kernel(const float* __restrict__ A,
                                 const float* __restrict__ W,
                                 const float* __restrict__ bias,
                                 float* __restrict__ C,
                                 int M, int K, int relu) {
  constexpr int Nc = NT * 16;
  extern __shared__ char smem_raw[];
  _Float16* sW = (_Float16*)smem_raw;

  const int tid = threadIdx.x;
  // stage W -> LDS in swizzled fragment order
  for (int i = tid; i < K * Nc; i += blockDim.x) {
    const int k = i / Nc;
    const int c = i - k * Nc;
    const int kc = k >> 5;
    const int r  = k & 31;
    const int j  = (r & 7) | ((r >> 4) << 3);       // half index within frag
    const int hiHalf = (r >> 3) & 1;                // lane-half select
    const int lane = (hiHalf << 4) | (c & 15);
    const int t = c >> 4;
    sW[(((kc * NT + t) << 5) + lane) * 16 + j] = (_Float16)W[i];
  }
  __syncthreads();

  const int wave  = tid >> 5;
  const int lane  = tid & 31;
  const int rbase = (blockIdx.x * 8 + wave) * 32;   // 2 M-tiles per wave
  if (rbase >= M) return;  // uniform per-wave exit: WMMA runs with full EXEC

  const bool hi  = lane >= 16;
  const int  col = lane & 15;
  int rowA0 = rbase + col;
  int rowA1 = rbase + 16 + col;
  if (rowA0 > M - 1) rowA0 = M - 1;                 // clamp: OOB rows never stored
  if (rowA1 > M - 1) rowA1 = M - 1;
  const float* __restrict__ ar0 = A + (size_t)rowA0 * K;
  const float* __restrict__ ar1 = A + (size_t)rowA1 * K;
  const int koff = hi ? 8 : 0;

  v8f acc0[NT], acc1[NT];
#pragma unroll
  for (int t = 0; t < NT; ++t)
#pragma unroll
    for (int j = 0; j < 8; ++j) { acc0[t][j] = 0.0f; acc1[t][j] = 0.0f; }

  const int chunks = K >> 5;
  for (int kc = 0; kc < chunks; ++kc) {
    const int k0 = (kc << 5) + koff;
    // A fragments: 2 rows x (two contiguous 8-float runs) as float4 loads
    v16h a0, a1;
    {
      const float4 p0 = *(const float4*)(ar0 + k0);
      const float4 p1 = *(const float4*)(ar0 + k0 + 4);
      const float4 p2 = *(const float4*)(ar0 + k0 + 16);
      const float4 p3 = *(const float4*)(ar0 + k0 + 20);
      const float4 q0 = *(const float4*)(ar1 + k0);
      const float4 q1 = *(const float4*)(ar1 + k0 + 4);
      const float4 q2 = *(const float4*)(ar1 + k0 + 16);
      const float4 q3 = *(const float4*)(ar1 + k0 + 20);
      a0[0]=(_Float16)p0.x; a0[1]=(_Float16)p0.y; a0[2]=(_Float16)p0.z; a0[3]=(_Float16)p0.w;
      a0[4]=(_Float16)p1.x; a0[5]=(_Float16)p1.y; a0[6]=(_Float16)p1.z; a0[7]=(_Float16)p1.w;
      a0[8]=(_Float16)p2.x; a0[9]=(_Float16)p2.y; a0[10]=(_Float16)p2.z; a0[11]=(_Float16)p2.w;
      a0[12]=(_Float16)p3.x; a0[13]=(_Float16)p3.y; a0[14]=(_Float16)p3.z; a0[15]=(_Float16)p3.w;
      a1[0]=(_Float16)q0.x; a1[1]=(_Float16)q0.y; a1[2]=(_Float16)q0.z; a1[3]=(_Float16)q0.w;
      a1[4]=(_Float16)q1.x; a1[5]=(_Float16)q1.y; a1[6]=(_Float16)q1.z; a1[7]=(_Float16)q1.w;
      a1[8]=(_Float16)q2.x; a1[9]=(_Float16)q2.y; a1[10]=(_Float16)q2.z; a1[11]=(_Float16)q2.w;
      a1[12]=(_Float16)q3.x; a1[13]=(_Float16)q3.y; a1[14]=(_Float16)q3.z; a1[15]=(_Float16)q3.w;
    }
#pragma unroll
    for (int t = 0; t < NT; ++t) {
      const v16h b = *(const v16h*)(sW + ((((kc * NT + t) << 5) + lane) << 4));
      acc0[t] = __builtin_amdgcn_wmma_f32_16x16x32_f16(
          false, a0, false, b, (short)0, acc0[t], false, false);
      acc1[t] = __builtin_amdgcn_wmma_f32_16x16x32_f16(
          false, a1, false, b, (short)0, acc1[t], false, false);
    }
  }

#pragma unroll
  for (int t = 0; t < NT; ++t) {
    const int cc = t * 16 + col;
    float bv = bias ? bias[cc] : 0.0f;
#pragma unroll
    for (int i = 0; i < 8; ++i) {
      const int mrow = hi ? 8 + i : i;
      int r = rbase + mrow;
      if (r < M) {
        float v = acc0[t][i] + bv;
        if (relu) v = fmaxf(v, 0.0f);
        C[(size_t)r * Nc + cc] = v;
      }
      r = rbase + 16 + mrow;
      if (r < M) {
        float v = acc1[t][i] + bv;
        if (relu) v = fmaxf(v, 0.0f);
        C[(size_t)r * Nc + cc] = v;
      }
    }
  }
}

// ------------------------------------------ edge gather/scale/scatter-add
// 16 threads (= Hdim/4 float4 chunks) per edge.
__global__ void edge_scatter_kernel(const int* __restrict__ ei,
                                    const float* __restrict__ dis,
                                    const float* __restrict__ Hb,
                                    float* __restrict__ ACC,
                                    int E, int Hdim, int shiftPer, int maskPer) {
  const int idx = blockIdx.x * blockDim.x + threadIdx.x;
  const int e = idx >> shiftPer;
  if (e >= E) return;
  const int c = (idx & maskPer) << 2;
  const int s = ei[e];
  const int d = ei[E + e];
  const float nr = dis[s] * dis[d];
  const float4 v = *(const float4*)(Hb + (size_t)s * Hdim + c);
  float* dp = ACC + (size_t)d * Hdim + c;
  atomicAdd(dp + 0, v.x * nr);
  atomicAdd(dp + 1, v.y * nr);
  atomicAdd(dp + 2, v.z * nr);
  atomicAdd(dp + 3, v.w * nr);
}

// out = relu(ACC + H*dis^2 + b); XC = out; HS (+)= out
__global__ void finalize_kernel(const float* __restrict__ ACC,
                                const float* __restrict__ Hb,
                                const float* __restrict__ dis,
                                const float* __restrict__ bias,
                                float* __restrict__ XC, float* __restrict__ HS,
                                int N, int shiftH, int maskH, int first) {
  const int idx = blockIdx.x * blockDim.x + threadIdx.x;
  const int n = idx >> shiftH;
  if (n >= N) return;
  const int f = idx & maskH;
  const float dn = dis[n];
  float v = ACC[idx] + Hb[idx] * (dn * dn) + bias[f];
  v = fmaxf(v, 0.0f);
  XC[idx] = v;
  HS[idx] = first ? v : HS[idx] + v;
}

__global__ void pool_scatter_kernel(const float* __restrict__ HS,
                                    const int* __restrict__ batch,
                                    float* __restrict__ POOL,
                                    float* __restrict__ CNT,
                                    int N, int Hdim, int shiftH, int maskH) {
  const int idx = blockIdx.x * blockDim.x + threadIdx.x;
  const int n = idx >> shiftH;
  if (n >= N) return;
  const int f = idx & maskH;
  const int g = batch[n];
  atomicAdd(&POOL[(size_t)g * Hdim + f], HS[idx]);
  if (f == 0) atomicAdd(&CNT[g], 1.0f);
}

__global__ void pool_divide_kernel(float* __restrict__ POOL,
                                   const float* __restrict__ CNT,
                                   int G, int shiftH) {
  const int idx = blockIdx.x * blockDim.x + threadIdx.x;
  const int g = idx >> shiftH;
  if (g >= G) return;
  POOL[idx] /= fmaxf(CNT[g], 1.0f);
}

// ---------------------------------------------------------------- launcher
extern "C" void kernel_launch(void* const* d_in, const int* in_sizes, int n_in,
                              void* d_out, int out_size, void* d_ws, size_t ws_size,
                              hipStream_t stream) {
  (void)n_in; (void)ws_size;
  const float* x    = (const float*)d_in[0];
  const int*   ei   = (const int*)d_in[1];
  const int*   batch= (const int*)d_in[2];
  const float* W1   = (const float*)d_in[3];
  const float* b1   = (const float*)d_in[4];
  const float* W2   = (const float*)d_in[5];
  const float* b2   = (const float*)d_in[6];
  const float* W3   = (const float*)d_in[7];
  const float* b3   = (const float*)d_in[8];
  const float* Wp1  = (const float*)d_in[9];
  const float* bp1  = (const float*)d_in[10];
  const float* Wp2  = (const float*)d_in[11];
  const float* bp2  = (const float*)d_in[12];
  float* out = (float*)d_out;

  const int N  = in_sizes[2];           // 50000 nodes
  const int E  = in_sizes[1] / 2;       // 800000 edges
  const int K0 = in_sizes[0] / N;       // 128 input features
  const int H  = in_sizes[4];           // 64 hidden
  const int HO = in_sizes[12];          // 32 output features
  const int G  = out_size / HO;         // 2500 graphs

  const int shiftH  = __builtin_ctz(H);       // 6
  const int maskH   = H - 1;
  const int shiftPer= __builtin_ctz(H / 4);   // 4 (float4 chunks/edge)
  const int maskPer = (H / 4) - 1;

  // workspace layout (floats), each region 256B aligned
  auto alignup = [](size_t f) { return (f + 63) & ~(size_t)63; };
  float* ws   = (float*)d_ws;
  size_t off  = 0;
  float* deg  = ws + off; off += alignup((size_t)N);           // deg -> dis in place
  float* Hb   = ws + off; off += alignup((size_t)N * H);
  float* ACC  = ws + off; off += alignup((size_t)N * H);
  float* XC   = ws + off; off += alignup((size_t)N * H);
  float* HS   = ws + off; off += alignup((size_t)N * H);
  float* POOL = ws + off; off += alignup((size_t)G * H);
  float* CNT  = ws + off; off += alignup((size_t)G);
  float* Z1   = ws + off; off += alignup((size_t)G * H);

  const int TPB = 256;
  auto nblk = [](long long n, int t) { return (int)((n + t - 1) / t); };

  // --- normalization: deg = 1 + incoming count; dis = rsqrt(deg)
  fill_kernel<<<nblk(N, TPB), TPB, 0, stream>>>(deg, 1.0f, N);
  degree_kernel<<<nblk(E, TPB), TPB, 0, stream>>>(ei, deg, E);
  rsqrt_kernel<<<nblk(N, TPB), TPB, 0, stream>>>(deg, N);
  const float* dis = deg;

  // --- 3 GCN layers
  const float* Wl[3] = {W1, W2, W3};
  const float* bl[3] = {b1, b2, b3};
  const float* Ain = x;
  int Kin = K0;
  for (int l = 0; l < 3; ++l) {
    size_t lds = (size_t)Kin * H * sizeof(_Float16);
    wmma_gemm_kernel<4><<<nblk(N, 256), TPB, lds, stream>>>(
        Ain, Wl[l], nullptr, Hb, N, Kin, 0);
    fill_kernel<<<nblk((long long)N * H, TPB), TPB, 0, stream>>>(ACC, 0.0f, N * H);
    edge_scatter_kernel<<<nblk((long long)E * (H / 4), TPB), TPB, 0, stream>>>(
        ei, dis, Hb, ACC, E, H, shiftPer, maskPer);
    finalize_kernel<<<nblk((long long)N * H, TPB), TPB, 0, stream>>>(
        ACC, Hb, dis, bl[l], XC, HS, N, shiftH, maskH, l == 0);
    Ain = XC;
    Kin = H;
  }

  // --- global mean pool over HS = x1+x2+x3
  fill_kernel<<<nblk((long long)G * H, TPB), TPB, 0, stream>>>(POOL, 0.0f, G * H);
  fill_kernel<<<nblk(G, TPB), TPB, 0, stream>>>(CNT, 0.0f, G);
  pool_scatter_kernel<<<nblk((long long)N * H, TPB), TPB, 0, stream>>>(
      HS, batch, POOL, CNT, N, H, shiftH, maskH);
  pool_divide_kernel<<<nblk((long long)G * H, TPB), TPB, 0, stream>>>(POOL, CNT, G, shiftH);

  // --- MLP head: relu(pooled@Wp1+bp1) @ Wp2 + bp2
  wmma_gemm_kernel<4><<<nblk(G, 256), TPB, (size_t)H * H * sizeof(_Float16), stream>>>(
      POOL, Wp1, bp1, Z1, G, H, 1);
  wmma_gemm_kernel<2><<<nblk(G, 256), TPB, (size_t)H * HO * sizeof(_Float16), stream>>>(
      Z1, Wp2, bp2, out, G, H, 0);
}
